// GraphSageLayerEdgeReprFeat_39049842655621
// MI455X (gfx1250) — matrix operand order repaired and checked by reading, not compiled, not run on hardware
//
#include <hip/hip_runtime.h>

typedef __attribute__((ext_vector_type(16))) _Float16 v16h;
typedef __attribute__((ext_vector_type(8)))  float    v8f;

#define F 128
#define TILE_M 16

// ---------------------------------------------------------------------------
// Fragment loaders (wave32, V_WMMA_F32_16X16X32_F16; contiguous k order chosen here, same for A and B)
// A: 16x32 f16. lane L (0-15): row M=L, K in {0..15}; lane L+16: row M=L,
//    K in {16..31}. VGPR p holds K-pair: K = hi8*16 + p*2.
// ---------------------------------------------------------------------------
__device__ __forceinline__ v16h load_a_frag_lds(const _Float16* hs, int lane, int kbase) {
    union { v16h v; unsigned u[8]; } f;
    const int m   = lane & 15;
    const int hi8 = (lane >> 4) & 1;
#pragma unroll
    for (int p = 0; p < 8; ++p) {
        int K = kbase + (hi8 << 4) + (p << 1);
        f.u[p] = *(const unsigned*)(hs + m * F + K);
    }
    return f.v;
}

// B: 32x16 f16. lane L (0-15): col N=L, K = 0..15 over VGPR halves; lane L+16:
// col N=L, K = 16..31. Weights stored transposed [n][k] -> 16 contiguous halfs.
__device__ __forceinline__ v16h load_b_frag(const _Float16* Wt, int ldk, int ncol,
                                            int lane, int kbase) {
    const int n    = ncol + (lane & 15);
    const int koff = kbase + (((lane >> 4) & 1) << 4);
    return *(const v16h*)(Wt + (size_t)n * ldk + koff);
}

// ---------------------------------------------------------------------------
// One-shot weight convert+transpose: WA/WB/WC (128x128) and Wap (256x128)
// Wt[n][k] layout so B-fragments are contiguous 32B loads.
// ---------------------------------------------------------------------------
__global__ void convert_weights_kernel(const float* __restrict__ WA,
                                       const float* __restrict__ WB,
                                       const float* __restrict__ WC,
                                       const float* __restrict__ Wap,
                                       _Float16* __restrict__ WAt,
                                       _Float16* __restrict__ WBt,
                                       _Float16* __restrict__ WCt,
                                       _Float16* __restrict__ Wapt) {
    int i = blockIdx.x * blockDim.x + threadIdx.x;
    if (i < 3 * F * F) {
        int mat = i / (F * F), j = i % (F * F);
        int k = j / F, n = j % F;
        const float* W  = (mat == 0) ? WA  : (mat == 1) ? WB  : WC;
        _Float16*    Wt = (mat == 0) ? WAt : (mat == 1) ? WBt : WCt;
        Wt[(size_t)n * F + k] = (_Float16)W[(size_t)k * F + n];
    } else if (i < 5 * F * F) {
        int j = i - 3 * F * F;
        int k = j / F, n = j % F;                 // k in 0..255
        Wapt[(size_t)n * (2 * F) + k] = (_Float16)Wap[(size_t)k * F + n];
    }
}

__global__ void zero_u32_kernel(unsigned* __restrict__ p, int n) {
    int i = blockIdx.x * blockDim.x + threadIdx.x;
    if (i < n) p[i] = 0u;
}

// ---------------------------------------------------------------------------
// Kernel 1: Ah = h@WA + bA, Bh = h@WB + bB  (stored fp16; L2 resident tables)
// 256 threads = 8 waves; wave w -> output cols [16w,16w+16); 16 node rows/block.
// Branch-free: out-of-range rows are clamped (duplicates write identical data).
// ---------------------------------------------------------------------------
__global__ void node_linear_kernel(const float* __restrict__ h,
                                   const _Float16* __restrict__ WAt,
                                   const _Float16* __restrict__ WBt,
                                   const float* __restrict__ bA,
                                   const float* __restrict__ bB,
                                   _Float16* __restrict__ Ah,
                                   _Float16* __restrict__ Bh, int N) {
    __shared__ __align__(16) _Float16 hs[TILE_M * F];

    const int tid  = threadIdx.x;
    const int row0 = blockIdx.x * TILE_M;

    {   // stage 16x128 f32 tile -> f16 LDS (8 elems/thread, vectorized)
        int base = tid * 8;
        int r = base >> 7, cc = base & 127;
        int row = min(row0 + r, N - 1);
        const float4* gp = (const float4*)(h + (size_t)row * F + cc);
        float4 a = gp[0], b = gp[1];
        _Float16* s = hs + r * F + cc;
        s[0] = (_Float16)a.x; s[1] = (_Float16)a.y; s[2] = (_Float16)a.z; s[3] = (_Float16)a.w;
        s[4] = (_Float16)b.x; s[5] = (_Float16)b.y; s[6] = (_Float16)b.z; s[7] = (_Float16)b.w;
    }
    __syncthreads();

    const int lane = tid & 31;
    const int colbase = (tid >> 5) * 16;

    v8f accA = {}; v8f accB = {};
#pragma unroll
    for (int kb = 0; kb < 4; ++kb) {
        const int kbase = kb * 32;
        v16h a  = load_a_frag_lds(hs, lane, kbase);
        v16h bA_ = load_b_frag(WAt, F, colbase, lane, kbase);
        v16h bB_ = load_b_frag(WBt, F, colbase, lane, kbase);
        accA = __builtin_amdgcn_wmma_f32_16x16x32_f16(false, a, false, bA_, (short)0, accA, false, false);
        accB = __builtin_amdgcn_wmma_f32_16x16x32_f16(false, a, false, bB_, (short)0, accB, false, false);
    }

    const int n   = colbase + (lane & 15);
    const int hi8 = (lane >> 4) & 1;
    const float ba = bA[n], bb = bB[n];
#pragma unroll
    for (int r = 0; r < 8; ++r) {
        int row = min(row0 + r + hi8 * 8, N - 1);
        Ah[(size_t)row * F + n] = (_Float16)(accA[r] + ba);
        Bh[(size_t)row * F + n] = (_Float16)(accB[r] + bb);
    }
}

// ---------------------------------------------------------------------------
// Kernel 2: edges. Ce = e@WC + bC (WMMA); e_ij = Ce + Bh[src] + Bh[dst];
// m = relu(sigmoid(e_ij)*Ah[src]); atomicMax bits into c; e_out = e + relu(e_ij)
// e_out repacked through LDS -> coalesced b128 stores (dominant HBM stream).
// ---------------------------------------------------------------------------
__global__ void edge_kernel(const float* __restrict__ e,
                            const int* __restrict__ src,
                            const int* __restrict__ dst,
                            const _Float16* __restrict__ WCt,
                            const float* __restrict__ bC,
                            const _Float16* __restrict__ Ah,
                            const _Float16* __restrict__ Bh,
                            unsigned* __restrict__ cbuf,
                            float* __restrict__ eout, int E) {
    __shared__ __align__(16) float    es[TILE_M * F];
    __shared__ __align__(16) _Float16 esh[TILE_M * F];
    __shared__ int sidx[TILE_M], didx[TILE_M];

    const int tid = threadIdx.x;
    const int e0  = blockIdx.x * TILE_M;

    {
        int base = tid * 8;
        int r = base >> 7, cc = base & 127;
        int eid = min(e0 + r, E - 1);
        const float4* gp = (const float4*)(e + (size_t)eid * F + cc);
        float4 a = gp[0], b = gp[1];
        float*    sf = es  + r * F + cc;
        _Float16* sh = esh + r * F + cc;
        sf[0]=a.x; sf[1]=a.y; sf[2]=a.z; sf[3]=a.w; sf[4]=b.x; sf[5]=b.y; sf[6]=b.z; sf[7]=b.w;
        sh[0]=(_Float16)a.x; sh[1]=(_Float16)a.y; sh[2]=(_Float16)a.z; sh[3]=(_Float16)a.w;
        sh[4]=(_Float16)b.x; sh[5]=(_Float16)b.y; sh[6]=(_Float16)b.z; sh[7]=(_Float16)b.w;
    }
    if (tid < TILE_M) {
        int ei = min(e0 + tid, E - 1);
        sidx[tid] = src[ei];
        didx[tid] = dst[ei];
    }
    __syncthreads();

    // Prefetch the 48 gathered rows (256B each = 2 cachelines) so the L2
    // round-trips overlap the WMMA loop below. (global_prefetch_b8 path)
    if (tid < 48) {
        int m = tid & 15, sel = tid >> 4;
        int rowi = (sel == 2) ? didx[m] : sidx[m];
        const _Float16* p = (sel == 0) ? (Ah + (size_t)rowi * F) : (Bh + (size_t)rowi * F);
        __builtin_prefetch(p, 0, 1);
        __builtin_prefetch((const char*)p + 128, 0, 1);
    }

    const int lane = tid & 31;
    const int colbase = (tid >> 5) * 16;

    v8f acc = {};
#pragma unroll
    for (int kb = 0; kb < 4; ++kb) {
        const int kbase = kb * 32;
        v16h a = load_a_frag_lds(esh, lane, kbase);
        v16h b = load_b_frag(WCt, F, colbase, lane, kbase);
        acc = __builtin_amdgcn_wmma_f32_16x16x32_f16(false, a, false, b, (short)0, acc, false, false);
    }

    const int n   = colbase + (lane & 15);
    const int hi8 = (lane >> 4) & 1;
    const float bc = bC[n];
#pragma unroll
    for (int r = 0; r < 8; ++r) {
        const int m = r + hi8 * 8;
        const int s = sidx[m], d = didx[m];
        float eij = acc[r] + bc + (float)Bh[(size_t)s * F + n] + (float)Bh[(size_t)d * F + n];
        float ah  = (float)Ah[(size_t)s * F + n];
        float sg  = 1.0f / (1.0f + __expf(-eij));
        float mv  = fmaxf(sg * ah, 0.0f);              // >= 0 -> uint max == float max
        atomicMax(&cbuf[(size_t)d * F + n], __float_as_uint(mv));
        es[m * F + n] += fmaxf(eij, 0.0f);             // in-place residual (bijective map)
    }
    __syncthreads();

    {   // coalesced b128 stores of e_new
        int base = tid * 8;
        int r = base >> 7, cc = base & 127;
        int eid = min(e0 + r, E - 1);
        const float4* sp = (const float4*)(es + r * F + cc);
        float4* gp = (float4*)(eout + (size_t)eid * F + cc);
        gp[0] = sp[0]; gp[1] = sp[1];
    }
}

// ---------------------------------------------------------------------------
// Kernel 3: bundle = [h, c] @ Wap + bap; L2-normalize rows; relu; h_out = h + .
// ---------------------------------------------------------------------------
__global__ void node_apply_kernel(const float* __restrict__ h,
                                  const float* __restrict__ cbuf,    // fp32 view of c
                                  const _Float16* __restrict__ Wapt, // [n][k], k=0..255
                                  const float* __restrict__ bap,
                                  float* __restrict__ hout, int N) {
    __shared__ __align__(16) float    hsf[TILE_M * F];
    __shared__ __align__(16) _Float16 hsh[TILE_M * F];
    __shared__ __align__(16) _Float16 csh[TILE_M * F];
    __shared__ float nrm[TILE_M];

    const int tid  = threadIdx.x;
    const int row0 = blockIdx.x * TILE_M;

    {
        int base = tid * 8;
        int r = base >> 7, cc = base & 127;
        int row = min(row0 + r, N - 1);
        const float4* gh = (const float4*)(h    + (size_t)row * F + cc);
        const float4* gc = (const float4*)(cbuf + (size_t)row * F + cc);
        float4 a = gh[0], b = gh[1], ca = gc[0], cb = gc[1];
        float*    sf = hsf + r * F + cc;
        _Float16* sh = hsh + r * F + cc;
        _Float16* sc = csh + r * F + cc;
        sf[0]=a.x; sf[1]=a.y; sf[2]=a.z; sf[3]=a.w; sf[4]=b.x; sf[5]=b.y; sf[6]=b.z; sf[7]=b.w;
        sh[0]=(_Float16)a.x; sh[1]=(_Float16)a.y; sh[2]=(_Float16)a.z; sh[3]=(_Float16)a.w;
        sh[4]=(_Float16)b.x; sh[5]=(_Float16)b.y; sh[6]=(_Float16)b.z; sh[7]=(_Float16)b.w;
        sc[0]=(_Float16)ca.x; sc[1]=(_Float16)ca.y; sc[2]=(_Float16)ca.z; sc[3]=(_Float16)ca.w;
        sc[4]=(_Float16)cb.x; sc[5]=(_Float16)cb.y; sc[6]=(_Float16)cb.z; sc[7]=(_Float16)cb.w;
    }
    if (tid < TILE_M) nrm[tid] = 0.0f;
    __syncthreads();

    const int lane = tid & 31;
    const int colbase = (tid >> 5) * 16;

    v8f acc = {};
#pragma unroll
    for (int kb = 0; kb < 4; ++kb) {            // K 0..127 : h part
        const int kbase = kb * 32;
        v16h a = load_a_frag_lds(hsh, lane, kbase);
        v16h b = load_b_frag(Wapt, 2 * F, colbase, lane, kbase);
        acc = __builtin_amdgcn_wmma_f32_16x16x32_f16(false, a, false, b, (short)0, acc, false, false);
    }
#pragma unroll
    for (int kb = 0; kb < 4; ++kb) {            // K 128..255 : c part
        const int kbase = kb * 32;
        v16h a = load_a_frag_lds(csh, lane, kbase);
        v16h b = load_b_frag(Wapt, 2 * F, colbase, lane, F + kbase);
        acc = __builtin_amdgcn_wmma_f32_16x16x32_f16(false, a, false, b, (short)0, acc, false, false);
    }

    const int n   = colbase + (lane & 15);
    const int hi8 = (lane >> 4) & 1;
    const float bb = bap[n];
    float vals[8];
#pragma unroll
    for (int r = 0; r < 8; ++r) {
        vals[r] = acc[r] + bb;
        float s = vals[r] * vals[r];
        s += __shfl_xor(s, 1, 32);
        s += __shfl_xor(s, 2, 32);
        s += __shfl_xor(s, 4, 32);
        s += __shfl_xor(s, 8, 32);              // stays within each 16-lane half
        if ((lane & 15) == 0) atomicAdd(&nrm[r + hi8 * 8], s);
    }
    __syncthreads();

#pragma unroll
    for (int r = 0; r < 8; ++r) {
        const int m = r + hi8 * 8;
        float denom = fmaxf(sqrtf(nrm[m]), 1e-12f);
        float hn    = fmaxf(vals[r] / denom, 0.0f);
        hsf[m * F + n] += hn;                   // in-place residual (bijective map)
    }
    __syncthreads();

    {   // coalesced b128 stores of h_new
        int base = tid * 8;
        int r = base >> 7, cc = base & 127;
        int row = min(row0 + r, N - 1);
        const float4* sp = (const float4*)(hsf + r * F + cc);
        float4* gp = (float4*)(hout + (size_t)row * F + cc);
        gp[0] = sp[0]; gp[1] = sp[1];
    }
}

// ---------------------------------------------------------------------------
extern "C" void kernel_launch(void* const* d_in, const int* in_sizes, int n_in,
                              void* d_out, int out_size, void* d_ws, size_t ws_size,
                              hipStream_t stream) {
    const float* h   = (const float*)d_in[0];
    const float* e   = (const float*)d_in[1];
    const int*   src = (const int*)d_in[2];
    const int*   dst = (const int*)d_in[3];
    const float* WA  = (const float*)d_in[4];
    const float* bA  = (const float*)d_in[5];
    const float* WB  = (const float*)d_in[6];
    const float* bB  = (const float*)d_in[7];
    const float* WC  = (const float*)d_in[8];
    const float* bC  = (const float*)d_in[9];
    const float* Wap = (const float*)d_in[10];
    const float* bap = (const float*)d_in[11];

    const int N = in_sizes[0] / F;
    const int E = in_sizes[2];

    // workspace carve-up
    char* ws = (char*)d_ws;
    _Float16* Ah   = (_Float16*)ws;  ws += (size_t)N * F * sizeof(_Float16);
    _Float16* Bh   = (_Float16*)ws;  ws += (size_t)N * F * sizeof(_Float16);
    unsigned* cbuf = (unsigned*)ws;  ws += (size_t)N * F * sizeof(unsigned);
    _Float16* WAt  = (_Float16*)ws;  ws += (size_t)F * F * sizeof(_Float16);
    _Float16* WBt  = (_Float16*)ws;  ws += (size_t)F * F * sizeof(_Float16);
    _Float16* WCt  = (_Float16*)ws;  ws += (size_t)F * F * sizeof(_Float16);
    _Float16* Wapt = (_Float16*)ws;  ws += (size_t)2 * F * F * sizeof(_Float16);

    float* hout = (float*)d_out;
    float* eout = hout + (size_t)N * F;

    // 1) weight convert/transpose (one launch) + zero segment-max accumulator
    convert_weights_kernel<<<(5 * F * F + 255) / 256, 256, 0, stream>>>(
        WA, WB, WC, Wap, WAt, WBt, WCt, Wapt);
    zero_u32_kernel<<<((size_t)N * F + 255) / 256, 256, 0, stream>>>(cbuf, N * F);

    // 2) node linears (Ah, Bh)
    node_linear_kernel<<<(N + TILE_M - 1) / TILE_M, 256, 0, stream>>>(
        h, WAt, WBt, bA, bB, Ah, Bh, N);

    // 3) edge stage: Ce + gather + gate + segment-max + e_out
    edge_kernel<<<(E + TILE_M - 1) / TILE_M, 256, 0, stream>>>(
        e, src, dst, WCt, bC, Ah, Bh, cbuf, eout, E);

    // 4) node apply: concat-GEMM + L2 normalize + relu + residual
    node_apply_kernel<<<(N + TILE_M - 1) / TILE_M, 256, 0, stream>>>(
        h, (const float*)cbuf, Wapt, bap, hout, N);
}
